// TTSAdapterAttention_34230889349794
// MI455X (gfx1250) — matrix operand order repaired and checked by reading, not compiled, not run on hardware
//
#include <hip/hip_runtime.h>

// ---------------- problem constants ----------------
#define B_ 2
#define T_ 2048
#define S_ 2048
#define QHID 2048
#define ENCHID 4096
#define H_ 32
#define KVH_ 8
#define HD_ 64

// ---------------- types ----------------
typedef __attribute__((ext_vector_type(8)))  __bf16 v8bf;
typedef __attribute__((ext_vector_type(16))) __bf16 v16bf;
typedef __attribute__((ext_vector_type(8)))  float  v8f;
typedef __attribute__((ext_vector_type(4)))  int    v4i;

union Frag {
  v16bf v;
  v8bf  h[2];
};

// native bf16 convert (RNE)
__device__ __forceinline__ unsigned short f2bf(float f) {
  __bf16 h = (__bf16)f;
  return __builtin_bit_cast(unsigned short, h);
}
__device__ __forceinline__ unsigned pk2(float a, float b) {
  return (unsigned)f2bf(a) | ((unsigned)f2bf(b) << 16);
}

__device__ __forceinline__ v8f wmma_bf16(const v16bf& a, const v16bf& b, const v8f& c) {
  return __builtin_amdgcn_wmma_f32_16x16x32_bf16(false, a, false, b, (short)0, c,
                                                 false, false);
}

// ---------------- async global -> LDS copy (16 B per lane) ----------------
#if defined(__AMDGCN__) && __has_builtin(__builtin_amdgcn_global_load_async_to_lds_b128)
#define ASYNC_LDS 1
typedef __attribute__((address_space(1))) v4i gv4i;   // global int4
typedef __attribute__((address_space(3))) v4i lv4i;   // LDS int4
__device__ __forceinline__ void cp16(const void* g, void* l) {
  __builtin_amdgcn_global_load_async_to_lds_b128((gv4i*)g, (lv4i*)l, 0, 0);
}
__device__ __forceinline__ void cp16_wait() {
#if __has_builtin(__builtin_amdgcn_s_wait_asynccnt)
  __builtin_amdgcn_s_wait_asynccnt(0);
#else
  asm volatile("s_wait_asynccnt 0x0" ::: "memory");
#endif
}
#else
#define ASYNC_LDS 0
__device__ __forceinline__ void cp16(const void* g, void* l) {
  *(uint4*)l = *(const uint4*)g;
}
__device__ __forceinline__ void cp16_wait() {}
#endif

// =====================================================================
// GEMM: Y[M,N] = X[M,K] * W[N,K]^T   (fp32 HBM -> bf16 LDS -> f32 WMMA acc)
// block 256 thr = 8 waves; tile 128x128; BK=32; register-staged pipeline.
// =====================================================================
#define GBM 128
#define GBN 128
#define GBK 32
#define GLD 40   // BK + 8 pad (ushorts)

template <bool OUT_BF16>
__global__ __launch_bounds__(256) void gemm_xwt(const float* __restrict__ X,
                                                const float* __restrict__ W,
                                                void* __restrict__ Yv,
                                                int M, int N, int K) {
  __shared__ unsigned short sA[GBM * GLD];
  __shared__ unsigned short sB[GBN * GLD];

  const int tid  = threadIdx.x;
  const int lane = tid & 31;
  const int wid  = tid >> 5;
  const int wm   = wid & 3;   // 0..3 : 32-row slice
  const int wn   = wid >> 2;  // 0..1 : 64-col slice
  const int m0   = blockIdx.y * GBM;
  const int n0   = blockIdx.x * GBN;

  v8f acc[2][4];
#pragma unroll
  for (int i = 0; i < 2; ++i)
#pragma unroll
    for (int j = 0; j < 4; ++j) acc[i][j] = (v8f)(0.f);

  const int half16 = lane >> 4;     // 0/1
  const int lidx   = lane & 15;
  const int klo    = half16 << 3;   // 0 / 8
  const int kh16   = half16 << 4;   // 0 / 16

  // staging coordinates: rows sr..sr+96 step 32; cols c4..c4+3
  const int sr = tid >> 3;          // 0..31
  const int c4 = (tid & 7) << 2;    // 0,4,...,28
  const float* Xp = X + (size_t)(m0 + sr) * K + c4;
  const float* Wp = W + (size_t)(n0 + sr) * K + c4;

  float4 ra[4], rb[4];

  auto gload = [&](int k0) {
#pragma unroll
    for (int i = 0; i < 4; ++i) {
      ra[i] = *(const float4*)(Xp + (size_t)(i * 32) * K + k0);
      rb[i] = *(const float4*)(Wp + (size_t)(i * 32) * K + k0);
    }
  };
  auto sstore = [&]() {
#pragma unroll
    for (int i = 0; i < 4; ++i) {
      uint2 pa, pb;
      pa.x = pk2(ra[i].x, ra[i].y);
      pa.y = pk2(ra[i].z, ra[i].w);
      pb.x = pk2(rb[i].x, rb[i].y);
      pb.y = pk2(rb[i].z, rb[i].w);
      *(uint2*)&sA[(sr + i * 32) * GLD + c4] = pa;
      *(uint2*)&sB[(sr + i * 32) * GLD + c4] = pb;
    }
  };

  gload(0);
  sstore();
  __syncthreads();

  for (int k0 = 0; k0 < K; k0 += GBK) {
    const bool nxt = (k0 + GBK) < K;
    if (nxt) gload(k0 + GBK);   // overlap next tile's HBM latency with WMMAs

    Frag a[2], b[4];
#pragma unroll
    for (int i = 0; i < 2; ++i) {
      int r = wm * 32 + i * 16 + lidx;
      a[i].h[0] = *(const v8bf*)(&sA[r * GLD + klo]);        // K = klo..klo+7
      a[i].h[1] = *(const v8bf*)(&sA[r * GLD + 16 + klo]);   // K = 16+klo..
    }
#pragma unroll
    for (int j = 0; j < 4; ++j) {
      int c = wn * 64 + j * 16 + lidx;
      b[j].h[0] = *(const v8bf*)(&sB[c * GLD + klo]);        // K = klo..klo+7
      b[j].h[1] = *(const v8bf*)(&sB[c * GLD + 16 + klo]);   // K = 16+klo..
    }
#pragma unroll
    for (int i = 0; i < 2; ++i)
#pragma unroll
      for (int j = 0; j < 4; ++j) acc[i][j] = wmma_bf16(a[i].v, b[j].v, acc[i][j]);

    if (nxt) {
      __syncthreads();
      sstore();
      __syncthreads();
    }
  }

  // ---- epilogue ----
  const int rbase = half16 << 3;
#pragma unroll
  for (int i = 0; i < 2; ++i)
#pragma unroll
    for (int j = 0; j < 4; ++j)
#pragma unroll
      for (int r = 0; r < 8; ++r) {
        int Mi = m0 + wm * 32 + i * 16 + rbase + r;
        int Ni = n0 + wn * 64 + j * 16 + lidx;
        if (OUT_BF16)
          ((unsigned short*)Yv)[(size_t)Mi * N + Ni] = f2bf(acc[i][j][r]);
        else
          ((float*)Yv)[(size_t)Mi * N + Ni] = acc[i][j][r];
      }
}

// =====================================================================
// Dual RoPE: y = x*cos + rotate_half(x)*sin ; fp32 [BL, nheads*64] -> bf16
// =====================================================================
__global__ void rope_bf16(const float* __restrict__ x, const float* __restrict__ c,
                          const float* __restrict__ s, unsigned short* __restrict__ y,
                          int total, int nheads) {
  int i = blockIdx.x * blockDim.x + threadIdx.x;
  if (i >= total) return;
  int d  = i & 63;
  int bl = (i >> 6) / nheads;  // b*L + l
  float xv  = x[i];
  float rot = (d < 32) ? -x[i + 32] : x[i - 32];
  y[i] = f2bf(xv * c[bl * 64 + d] + rot * s[bl * 64 + d]);
}

// =====================================================================
// V transpose: vb bf16 [B, S, KVH*64] -> vt bf16 [B, KVH, 64, S]
// (enables contiguous async copies of V^T rows in the attention kernel)
// =====================================================================
__global__ __launch_bounds__(256) void vtrans(const unsigned short* __restrict__ vb,
                                              unsigned short* __restrict__ vt) {
  __shared__ unsigned short tile[64 * 72];
  const int s0  = blockIdx.x * 64;
  const int bk  = blockIdx.y;          // b*KVH + kvh
  const int b   = bk >> 3;
  const int kvh = bk & 7;
  const int tid = threadIdx.x;
  const int r   = tid >> 3;            // 0..31
  const int g   = (tid & 7) << 3;      // 0,8,...,56
#pragma unroll
  for (int i = 0; i < 2; ++i) {
    int row = r + i * 32;              // s offset
    *(uint4*)&tile[row * 72 + g] =
        *(const uint4*)(vb + ((size_t)b * S_ + s0 + row) * (KVH_ * HD_) + kvh * HD_ + g);
  }
  __syncthreads();
#pragma unroll
  for (int i = 0; i < 2; ++i) {
    int drow = r + i * 32;             // d index
    union { uint4 q; unsigned short us[8]; } t;
#pragma unroll
    for (int j = 0; j < 8; ++j) t.us[j] = tile[(g + j) * 72 + drow];
    *(uint4*)(vt + (((size_t)b * KVH_ + kvh) * HD_ + drow) * S_ + s0 + g) = t.q;
  }
}

// =====================================================================
// Flash attention, GQA (group=4). Block: 64 queries of one (b,h).
// 128 thr = 4 waves, 16 queries each; S in 64-key chunks; K/V^T streamed
// into double-buffered LDS with async DMA (no VGPR staging).
// =====================================================================
#define ALD 72   // 64 + 8 pad (ushorts)

__global__ __launch_bounds__(128) void attn(const unsigned short* __restrict__ qb,
                                            const unsigned short* __restrict__ kb,
                                            const unsigned short* __restrict__ vt,
                                            const float* __restrict__ mask,
                                            float* __restrict__ ctx) {
  const int t0  = blockIdx.x * 64;
  const int h   = blockIdx.y;
  const int b   = blockIdx.z;
  const int kvh = h >> 2;

  __shared__ unsigned short sQ[64 * ALD];
  __shared__ unsigned short sK[2][64 * ALD];
  __shared__ unsigned short sVT[2][64 * ALD];       // [d][key]
  __shared__ unsigned short sP[4][16 * ALD];        // per-wave P tile

  const int tid    = threadIdx.x;
  const int lane   = tid & 31;
  const int wid    = tid >> 5;
  const int half16 = lane >> 4;
  const int lidx   = lane & 15;
  const int klo    = half16 << 3;
  const int kh16   = half16 << 4;
  const int rbase  = half16 << 3;

  // staging coordinates (64 rows x 8 uint4-groups, 4 per thread)
  const int qrow = tid >> 3;          // 0..15 base; +16*i
  const int qg   = (tid & 7) << 3;    // 0,8,...,56

  const unsigned short* kbp = kb + (size_t)b * S_ * (KVH_ * HD_) + kvh * HD_;
  const unsigned short* vtp = vt + (((size_t)b * KVH_ + kvh) * HD_) * S_;

  auto stageKV = [&](int s0, int bb) {
#pragma unroll
    for (int i = 0; i < 4; ++i) {
      int row = qrow + i * 16;
      cp16(kbp + (size_t)(s0 + row) * (KVH_ * HD_) + qg, &sK[bb][row * ALD + qg]);
      cp16(vtp + (size_t)row * S_ + s0 + qg,             &sVT[bb][row * ALD + qg]);
    }
  };

  // ---- stage Q tile + first K/V chunk ----
#pragma unroll
  for (int i = 0; i < 4; ++i) {
    int row = qrow + i * 16;
    *(uint4*)&sQ[row * ALD + qg] =
        *(const uint4*)(qb + ((size_t)b * T_ + t0 + row) * (H_ * HD_) + h * HD_ + qg);
  }
  stageKV(0, 0);
  cp16_wait();
  __syncthreads();

  // ---- per-wave Q fragments (held in VGPRs for whole kernel) ----
  Frag qf2[2];
#pragma unroll
  for (int w = 0; w < 2; ++w) {
    int r = wid * 16 + lidx;
    qf2[w].h[0] = *(const v8bf*)&sQ[r * ALD + w * 32 + klo];
    qf2[w].h[1] = *(const v8bf*)&sQ[r * ALD + w * 32 + 16 + klo];
  }

  float mrow[8], lrow[8];
  v8f oacc[4];
#pragma unroll
  for (int r = 0; r < 8; ++r) { mrow[r] = -1e30f; lrow[r] = 0.f; }
#pragma unroll
  for (int dt = 0; dt < 4; ++dt) oacc[dt] = (v8f)(0.f);

  for (int s0 = 0, bb = 0; s0 < S_; s0 += 64, bb ^= 1) {
    const bool nxt = (s0 + 64) < S_;
    // async-DMA next chunk into the other buffer (last read in the previous iter,
    // already barrier-separated) while this chunk computes
    if (nxt) stageKV(s0 + 64, bb ^ 1);

    const unsigned short* cK  = sK[bb];
    const unsigned short* cVT = sVT[bb];

    // ---- scores = Q @ K^T (4 key-subtiles x 2 K-halves) ----
    v8f sc[4];
#pragma unroll
    for (int st = 0; st < 4; ++st) sc[st] = (v8f)(0.f);
#pragma unroll
    for (int st = 0; st < 4; ++st) {
#pragma unroll
      for (int w = 0; w < 2; ++w) {
        Frag kf;
        int n = st * 16 + lidx;
        kf.h[0] = *(const v8bf*)&cK[n * ALD + w * 32 + klo];
        kf.h[1] = *(const v8bf*)&cK[n * ALD + w * 32 + 16 + klo];
        sc[st] = wmma_bf16(qf2[w].v, kf.v, sc[st]);
      }
    }
    // ---- scale + additive mask ----
#pragma unroll
    for (int st = 0; st < 4; ++st)
#pragma unroll
      for (int r = 0; r < 8; ++r) {
        int t = t0 + wid * 16 + rbase + r;
        int s = s0 + st * 16 + lidx;
        sc[st][r] = sc[st][r] * 0.125f + mask[((size_t)b * T_ + t) * S_ + s];
      }

    // ---- online softmax (row stats across 16-lane half-wave groups) ----
#pragma unroll
    for (int r = 0; r < 8; ++r) {
      float mx = fmaxf(fmaxf(sc[0][r], sc[1][r]), fmaxf(sc[2][r], sc[3][r]));
      mx = fmaxf(mx, __shfl_xor(mx, 1));
      mx = fmaxf(mx, __shfl_xor(mx, 2));
      mx = fmaxf(mx, __shfl_xor(mx, 4));
      mx = fmaxf(mx, __shfl_xor(mx, 8));
      float mnew = fmaxf(mrow[r], mx);
      float corr = __expf(mrow[r] - mnew);
      float rs = 0.f;
#pragma unroll
      for (int st = 0; st < 4; ++st) {
        float p = __expf(sc[st][r] - mnew);
        sP[wid][(rbase + r) * ALD + st * 16 + lidx] = f2bf(p);
        rs += p;
      }
      rs += __shfl_xor(rs, 1);
      rs += __shfl_xor(rs, 2);
      rs += __shfl_xor(rs, 4);
      rs += __shfl_xor(rs, 8);
      lrow[r] = lrow[r] * corr + rs;
      mrow[r] = mnew;
#pragma unroll
      for (int dt = 0; dt < 4; ++dt) oacc[dt][r] *= corr;
    }
    // wave-private LDS region: ensure P stores land before fragment reads
    asm volatile("s_wait_dscnt 0" ::: "memory");

    // ---- O += P @ V ----
#pragma unroll
    for (int dt = 0; dt < 4; ++dt) {
#pragma unroll
      for (int w = 0; w < 2; ++w) {
        Frag pf, vf;
        pf.h[0] = *(const v8bf*)&sP[wid][lidx * ALD + w * 32 + klo];
        pf.h[1] = *(const v8bf*)&sP[wid][lidx * ALD + w * 32 + 16 + klo];
        int dcol = dt * 16 + lidx;
        vf.h[0] = *(const v8bf*)&cVT[dcol * ALD + w * 32 + klo];
        vf.h[1] = *(const v8bf*)&cVT[dcol * ALD + w * 32 + 16 + klo];
        oacc[dt] = wmma_bf16(pf.v, vf.v, oacc[dt]);
      }
    }

    if (nxt) {
      cp16_wait();       // this wave's DMA into bb^1 has landed
      __syncthreads();   // publish to all waves; all reads of bb done
    }
  }

  // ---- finalize: O /= l ; write fp32 ctx [B,T,H*64] ----
#pragma unroll
  for (int dt = 0; dt < 4; ++dt)
#pragma unroll
    for (int r = 0; r < 8; ++r) {
      int t = t0 + wid * 16 + rbase + r;
      int d = dt * 16 + lidx;
      ctx[((size_t)b * T_ + t) * (H_ * HD_) + h * HD_ + d] = oacc[dt][r] / lrow[r];
    }
}

// =====================================================================
// launch
// =====================================================================
extern "C" void kernel_launch(void* const* d_in, const int* in_sizes, int n_in,
                              void* d_out, int out_size, void* d_ws, size_t ws_size,
                              hipStream_t stream) {
  const float* hs   = (const float*)d_in[0];   // [B,T,2048]
  const float* kv   = (const float*)d_in[1];   // [B,S,4096]
  const float* mask = (const float*)d_in[2];   // [B,1,T,S]
  const float* cosq = (const float*)d_in[3];   // [B,T,64]
  const float* sinq = (const float*)d_in[4];
  const float* ecos = (const float*)d_in[5];   // [B,S,64]
  const float* esin = (const float*)d_in[6];
  const float* Wq   = (const float*)d_in[7];   // [2048,2048]
  const float* Wk   = (const float*)d_in[8];   // [512,4096]
  const float* Wv   = (const float*)d_in[9];   // [512,4096]
  const float* Wo   = (const float*)d_in[10];  // [2048,2048]
  float* out = (float*)d_out;

  char* ws = (char*)d_ws;
  float*          qf  = (float*)(ws + 0);                   // 32 MB  q proj f32
  float*          kf  = (float*)(ws + 33554432);            //  8 MB  k proj f32
  unsigned short* vb  = (unsigned short*)(ws + 41943040);   //  4 MB  v bf16
  unsigned short* qb  = (unsigned short*)(ws + 46137344);   // 16 MB  q rope bf16
  unsigned short* kb  = (unsigned short*)(ws + 62914560);   //  4 MB  k rope bf16
  float*          ctx = (float*)(ws + 67108864);            // 32 MB  attn out f32
  unsigned short* vt  = (unsigned short*)(ws + 100663296);  //  4 MB  v^T bf16

  const int MQ = B_ * T_;  // 4096
  const int MS = B_ * S_;  // 4096

  // projections
  gemm_xwt<false><<<dim3(QHID / GBN, MQ / GBM), 256, 0, stream>>>(hs, Wq, qf, MQ, QHID, QHID);
  gemm_xwt<false><<<dim3((KVH_ * HD_) / GBN, MS / GBM), 256, 0, stream>>>(kv, Wk, kf, MS, KVH_ * HD_, ENCHID);
  gemm_xwt<true ><<<dim3((KVH_ * HD_) / GBN, MS / GBM), 256, 0, stream>>>(kv, Wv, vb, MS, KVH_ * HD_, ENCHID);

  // dual RoPE + V transpose
  {
    int totq = B_ * T_ * H_ * HD_;    // 8388608
    int totk = B_ * S_ * KVH_ * HD_;  // 2097152
    rope_bf16<<<(totq + 255) / 256, 256, 0, stream>>>(qf, cosq, sinq, qb, totq, H_);
    rope_bf16<<<(totk + 255) / 256, 256, 0, stream>>>(kf, ecos, esin, kb, totk, KVH_);
    vtrans<<<dim3(S_ / 64, B_ * KVH_), 256, 0, stream>>>(vb, vt);
  }

  // fused flash attention with GQA
  attn<<<dim3(T_ / 64, H_, B_), 128, 0, stream>>>(qb, kb, vt, mask, ctx);

  // output projection
  gemm_xwt<false><<<dim3(QHID / GBN, MQ / GBM), 256, 0, stream>>>(ctx, Wo, out, MQ, QHID, QHID);
}
